// CMRN_39685497815926
// MI455X (gfx1250) — matrix operand order repaired and checked by reading, not compiled, not run on hardware
//
#include <hip/hip_runtime.h>
#include <hip/hip_bf16.h>

// ---------------------------------------------------------------------------
// Types for CDNA5 WMMA
// ---------------------------------------------------------------------------
typedef __attribute__((ext_vector_type(16))) __bf16 v16bf;
typedef __attribute__((ext_vector_type(8)))  float  v8f;

// Native converts (round-to-nearest-even) — lowers to v_cvt_pk_bf16_f32
__device__ __forceinline__ unsigned pack2_bf16(float x, float y) {
  union { __bf16 h[2]; unsigned u; } c;
  c.h[0] = (__bf16)x; c.h[1] = (__bf16)y;
  return c.u;
}

__device__ __forceinline__ float gelu_tanh(float x) {
  float x3 = x * x * x;
  return 0.5f * x * (1.0f + tanhf(0.7978845608028654f * (x + 0.044715f * x3)));
}

// ---------------------------------------------------------------------------
// Tiled WMMA GEMM: out = epilogue(A[M,K] * B[K,N] + bias)
//   A, B f32 in global, converted to bf16 while staging to LDS.
//   Block tile 128x128, K-step 32, double-buffered LDS software pipeline:
//   global loads of tile s+1 (into regs) overlap WMMA on tile s.
//   A staged as [m][k] rows; B staged TRANSPOSED [n][k] with k-contiguous
//   stores (2 x ds_store_b128 per thread) so no bit-shuffling is needed.
//   256 threads = 8 waves; each wave: 32x64 sub-tile = 2x4 grid of
//   v_wmma_f32_16x16x32_bf16 accumulators.
// EPI: 0 = gelu(acc+bias)            -> out
//      1 = resid + acc + bias        -> out
//      2 = out += scale[m]*(acc+bias)   (routing-weighted expert combine)
//      3 = acc + bias                -> out
// ---------------------------------------------------------------------------
#define BM 128
#define BN 128
#define BKK 32
#define LDS_AS 40   // bf16 elems per A row in LDS (padded)
#define LDS_BS 40   // bf16 elems per B^T row in LDS (padded)

template <int EPI>
__global__ __launch_bounds__(256) void wmma_gemm_kernel(
    const float* __restrict__ A, const float* __restrict__ Bw,
    const float* __restrict__ bias, float* __restrict__ out,
    const float* __restrict__ resid, const float* __restrict__ scale,
    int M, int N, int Kd, int scale_stride, int scale_off) {
  alignas(16) __shared__ unsigned short lA[2][BM * LDS_AS];
  alignas(16) __shared__ unsigned short lB[2][BN * LDS_BS];  // B^T: [n][k]

  const int tid  = threadIdx.x;
  const int lane = tid & 31;
  const int wave = tid >> 5;     // 0..7
  const int wm   = wave >> 1;    // 0..3 : 32-row strip
  const int wn   = wave & 1;     // 0..1 : 64-col strip
  const int halfsel = lane >> 4; // 0 or 1
  const int lr      = lane & 15;

  const int rowBase = blockIdx.y * BM;
  const int colBase = blockIdx.x * BN;

  // ---- A staging coords: thread -> 4 float4 (row r, 4 consecutive k) ------
  const float* aPtr[4]; int aOff[4];
#pragma unroll
  for (int i = 0; i < 4; ++i) {
    int v  = i * 256 + tid;          // 1024 float4 loads for A tile
    int r  = v >> 3;                 // 8 float4 per 32-elem row
    int c4 = (v & 7) * 4;
    aOff[i] = r * LDS_AS + c4;
    aPtr[i] = A + (size_t)(rowBase + r) * Kd + c4;
  }
  // ---- B staging coords: thread -> column n, 16 consecutive k -------------
  const int bn = tid & 127;          // column within tile
  const int kg = (tid >> 7) * 16;    // k-range start (0 or 16)
  const float* bPtr = Bw + (size_t)kg * N + colBase + bn;
  const int bOff = bn * LDS_BS + kg; // 16 contiguous bf16, 16B aligned

  v8f acc[2][4];
  const v8f vzero = {};
#pragma unroll
  for (int i = 0; i < 2; ++i)
#pragma unroll
    for (int j = 0; j < 4; ++j) acc[i][j] = vzero;

  float4 ra[4];
  float  rbv[16];

  // ---- pipeline prologue: fetch tile 0 into regs, commit to LDS buf 0 -----
#pragma unroll
  for (int i = 0; i < 4; ++i) { ra[i] = *(const float4*)aPtr[i]; aPtr[i] += BKK; }
#pragma unroll
  for (int j = 0; j < 16; ++j) rbv[j] = bPtr[(size_t)j * N];
  bPtr += (size_t)BKK * N;
#pragma unroll
  for (int i = 0; i < 4; ++i) {
    unsigned* d = (unsigned*)&lA[0][aOff[i]];
    d[0] = pack2_bf16(ra[i].x, ra[i].y);
    d[1] = pack2_bf16(ra[i].z, ra[i].w);
  }
  {
    unsigned* d = (unsigned*)&lB[0][bOff];
#pragma unroll
    for (int p = 0; p < 8; ++p) d[p] = pack2_bf16(rbv[2 * p], rbv[2 * p + 1]);
  }
  __syncthreads();

  const int nsteps = Kd / BKK;
  for (int s = 0; s < nsteps; ++s) {
    const int cur  = s & 1;
    const bool more = (s + 1 < nsteps);

    // ---- issue next tile's global loads (latency hidden behind WMMA) -----
    if (more) {
#pragma unroll
      for (int i = 0; i < 4; ++i) { ra[i] = *(const float4*)aPtr[i]; aPtr[i] += BKK; }
#pragma unroll
      for (int j = 0; j < 16; ++j) rbv[j] = bPtr[(size_t)j * N];
      bPtr += (size_t)BKK * N;
      __builtin_prefetch(aPtr[0], 0, 1);   // gfx1250 global_prefetch_b8
      __builtin_prefetch(bPtr, 0, 1);
    }

    // ---- build fragments per ISA VGPR layouts ----------------------------
    // A 16x32 bf16: lane<16 -> s[0..7]=K0..7, s[8..15]=K16..23 ; lane>=16 +8
    v16bf aF[2], bF[4];
#pragma unroll
    for (int ti = 0; ti < 2; ++ti) {
      int m  = wm * 32 + ti * 16 + lr;
      int kb = halfsel * 8;
      union { v16bf v; unsigned short sdat[16]; } u;
#pragma unroll
      for (int j = 0; j < 8; ++j) u.sdat[j]     = lA[cur][m * LDS_AS + kb + j];
#pragma unroll
      for (int j = 0; j < 8; ++j) u.sdat[8 + j] = lA[cur][m * LDS_AS + kb + 16 + j];
      aF[ti] = u.v;
    }
    // B 32x16 bf16: col=lane%16; lane<16 -> s[0..7]=K0..7, s[8..15]=K16..23 ; lane>=16 +8
#pragma unroll
    for (int tj = 0; tj < 4; ++tj) {
      int n  = wn * 64 + tj * 16 + lr;
      int kb = halfsel * 8;
      union { v16bf v; unsigned short sdat[16]; } u;
#pragma unroll
      for (int j = 0; j < 8; ++j) u.sdat[j]     = lB[cur][n * LDS_BS + kb + j];
#pragma unroll
      for (int j = 0; j < 8; ++j) u.sdat[8 + j] = lB[cur][n * LDS_BS + kb + 16 + j];
      bF[tj] = u.v;
    }

#pragma unroll
    for (int ti = 0; ti < 2; ++ti)
#pragma unroll
      for (int tj = 0; tj < 4; ++tj)
        acc[ti][tj] = __builtin_amdgcn_wmma_f32_16x16x32_bf16(
            false, aF[ti], false, bF[tj], (short)0, acc[ti][tj], false, false);

    // ---- commit next tile to the other LDS buffer ------------------------
    if (more) {
      const int nxt = cur ^ 1;
#pragma unroll
      for (int i = 0; i < 4; ++i) {
        unsigned* d = (unsigned*)&lA[nxt][aOff[i]];
        d[0] = pack2_bf16(ra[i].x, ra[i].y);
        d[1] = pack2_bf16(ra[i].z, ra[i].w);
      }
      unsigned* d = (unsigned*)&lB[nxt][bOff];
#pragma unroll
      for (int p = 0; p < 8; ++p) d[p] = pack2_bf16(rbv[2 * p], rbv[2 * p + 1]);
    }
    __syncthreads();
  }

  // ---- epilogue: C/D layout -> m = r + 8*halfsel, n = lr -------------------
#pragma unroll
  for (int ti = 0; ti < 2; ++ti)
#pragma unroll
    for (int tj = 0; tj < 4; ++tj)
#pragma unroll
      for (int r = 0; r < 8; ++r) {
        int m = rowBase + wm * 32 + ti * 16 + halfsel * 8 + r;
        int n = colBase + wn * 64 + tj * 16 + lr;
        size_t idx = (size_t)m * N + n;
        float v = acc[ti][tj][r] + bias[n];
        if (EPI == 0) {
          out[idx] = gelu_tanh(v);
        } else if (EPI == 1) {
          out[idx] = resid[idx] + v;
        } else if (EPI == 2) {
          float sc = scale[(size_t)m * scale_stride + scale_off];
          out[idx] += sc * v;
        } else {
          out[idx] = v;
        }
      }
}

// ---------------------------------------------------------------------------
// Embedding gather: x[t,:] = emb[ids[t],:]
// ---------------------------------------------------------------------------
__global__ __launch_bounds__(256) void embed_kernel(
    const int* __restrict__ ids, const float* __restrict__ emb,
    float* __restrict__ x, int D) {
  int t  = blockIdx.x;
  int id = ids[t];
  const float4* src = (const float4*)(emb + (size_t)id * D);
  float4*       dst = (float4*)(x + (size_t)t * D);
  for (int i = threadIdx.x; i < D / 4; i += 256) dst[i] = src[i];
}

// ---------------------------------------------------------------------------
// LayerNorm (one block per row)
// ---------------------------------------------------------------------------
__global__ __launch_bounds__(256) void ln_kernel(
    const float* __restrict__ x, const float* __restrict__ g,
    const float* __restrict__ b, float* __restrict__ out, int D) {
  __shared__ float red[256];
  int row = blockIdx.x;
  const float* xr = x + (size_t)row * D;
  float s = 0.f, s2 = 0.f;
  for (int d = threadIdx.x; d < D; d += 256) {
    float v = xr[d]; s += v; s2 += v * v;
  }
  red[threadIdx.x] = s; __syncthreads();
  for (int off = 128; off > 0; off >>= 1) {
    if (threadIdx.x < off) red[threadIdx.x] += red[threadIdx.x + off];
    __syncthreads();
  }
  float mean = red[0] / D; __syncthreads();
  red[threadIdx.x] = s2; __syncthreads();
  for (int off = 128; off > 0; off >>= 1) {
    if (threadIdx.x < off) red[threadIdx.x] += red[threadIdx.x + off];
    __syncthreads();
  }
  float var  = red[0] / D - mean * mean;
  float rstd = rsqrtf(var + 1e-5f);
  for (int d = threadIdx.x; d < D; d += 256)
    out[(size_t)row * D + d] = (xr[d] - mean) * rstd * g[d] + b[d];
}

// ---------------------------------------------------------------------------
// Router: r = x*Wr + br, top-2 renormalized softmax scattered to [t,E]
// One wave (32 lanes) per token; 8 tokens per block.
// ---------------------------------------------------------------------------
__global__ __launch_bounds__(256) void router_kernel(
    const float* __restrict__ x, const float* __restrict__ Wr,
    const float* __restrict__ br, float* __restrict__ rw,
    float* __restrict__ rw_out, int D) {
  const int lane = threadIdx.x & 31;
  const int wv   = threadIdx.x >> 5;
  const int t    = blockIdx.x * 8 + wv;
  float acc[8] = {0, 0, 0, 0, 0, 0, 0, 0};
  const float* xr = x + (size_t)t * D;
  for (int d = lane; d < D; d += 32) {
    float xv = xr[d];
    const float4 w0 = *(const float4*)(Wr + (size_t)d * 8);
    const float4 w1 = *(const float4*)(Wr + (size_t)d * 8 + 4);
    acc[0] += xv * w0.x; acc[1] += xv * w0.y;
    acc[2] += xv * w0.z; acc[3] += xv * w0.w;
    acc[4] += xv * w1.x; acc[5] += xv * w1.y;
    acc[6] += xv * w1.z; acc[7] += xv * w1.w;
  }
#pragma unroll
  for (int e = 0; e < 8; ++e)
#pragma unroll
    for (int off = 16; off > 0; off >>= 1)
      acc[e] += __shfl_down(acc[e], off, 32);
  if (lane == 0) {
    float lg[8];
#pragma unroll
    for (int e = 0; e < 8; ++e) lg[e] = acc[e] + br[e];
    int i1 = 0; float v1 = lg[0];
#pragma unroll
    for (int e = 1; e < 8; ++e)
      if (lg[e] > v1) { v1 = lg[e]; i1 = e; }
    int i2 = -1; float v2 = -3.4e38f;
#pragma unroll
    for (int e = 0; e < 8; ++e)
      if (e != i1 && lg[e] > v2) { v2 = lg[e]; i2 = e; }
    float ex = __expf(v2 - v1);
    float w1 = 1.f / (1.f + ex);
    float w2 = ex * w1;
    float o[8] = {0, 0, 0, 0, 0, 0, 0, 0};
    o[i1] = w1; o[i2] = w2;
#pragma unroll
    for (int e = 0; e < 8; ++e) {
      rw[(size_t)t * 8 + e]     = o[e];
      rw_out[(size_t)t * 8 + e] = o[e];
    }
  }
}

__global__ void zero_kernel(float* p, int n) {
  int i = blockIdx.x * blockDim.x + threadIdx.x;
  if (i < n) p[i] = 0.f;
}

// ---------------------------------------------------------------------------
extern "C" void kernel_launch(void* const* d_in, const int* in_sizes, int n_in,
                              void* d_out, int out_size, void* d_ws,
                              size_t ws_size, hipStream_t stream) {
  (void)in_sizes; (void)n_in; (void)out_size; (void)ws_size;
  const int Vv = 32000, D = 1024, DFF = 4096, E = 8, L = 2;
  const int M = 2 * 1024;  // B*S tokens

  const int*   ids   = (const int*)d_in[0];
  const float* emb   = (const float*)d_in[1];
  const float* lnS   = (const float*)d_in[2];
  const float* lnB   = (const float*)d_in[3];
  const float* t_w1  = (const float*)d_in[4];
  const float* t_b1  = (const float*)d_in[5];
  const float* t_w2  = (const float*)d_in[6];
  const float* t_b2  = (const float*)d_in[7];
  const float* Wr    = (const float*)d_in[8];
  const float* br    = (const float*)d_in[9];
  const float* e_w1  = (const float*)d_in[10];
  const float* e_b1  = (const float*)d_in[11];
  const float* e_w2  = (const float*)d_in[12];
  const float* e_b2  = (const float*)d_in[13];
  const float* Wl    = (const float*)d_in[14];
  const float* bl    = (const float*)d_in[15];

  float* x   = (float*)d_ws;
  float* hln = x + (size_t)M * D;
  float* h   = hln + (size_t)M * D;
  float* y   = h + (size_t)M * DFF;
  float* rw  = y + (size_t)M * D;

  float* logits = (float*)d_out;
  float* rw_out = logits + (size_t)M * Vv;

  dim3 blk(256);

  embed_kernel<<<M, blk, 0, stream>>>(ids, emb, x, D);

  for (int l = 0; l < L; ++l) {
    ln_kernel<<<M, blk, 0, stream>>>(x, lnS + (size_t)l * D,
                                     lnB + (size_t)l * D, hln, D);
    wmma_gemm_kernel<0><<<dim3(DFF / BN, M / BM), blk, 0, stream>>>(
        hln, t_w1 + (size_t)l * D * DFF, t_b1 + (size_t)l * DFF, h, nullptr,
        nullptr, M, DFF, D, 0, 0);
    wmma_gemm_kernel<1><<<dim3(D / BN, M / BM), blk, 0, stream>>>(
        h, t_w2 + (size_t)l * DFF * D, t_b2 + (size_t)l * D, x, x, nullptr, M,
        D, DFF, 0, 0);
  }

  router_kernel<<<M / 8, blk, 0, stream>>>(x, Wr, br, rw, rw_out, D);

  zero_kernel<<<(M * D + 255) / 256, blk, 0, stream>>>(y, M * D);

  for (int e = 0; e < E; ++e) {
    wmma_gemm_kernel<0><<<dim3(DFF / BN, M / BM), blk, 0, stream>>>(
        x, e_w1 + (size_t)e * D * DFF, e_b1 + (size_t)e * DFF, h, nullptr,
        nullptr, M, DFF, D, 0, 0);
    wmma_gemm_kernel<2><<<dim3(D / BN, M / BM), blk, 0, stream>>>(
        h, e_w2 + (size_t)e * DFF * D, e_b2 + (size_t)e * D, y, nullptr, rw, M,
        D, DFF, E, e);
  }

  wmma_gemm_kernel<3><<<dim3(Vv / BN, M / BM), blk, 0, stream>>>(
      y, Wl, bl, logits, nullptr, nullptr, M, Vv, D, 0, 0);
}
